// CompressedLinearFP16FP32_83081847374546
// MI455X (gfx1250) — matrix operand order repaired and checked by reading, not compiled, not run on hardware
//
#include <hip/hip_runtime.h>

// out[8, 11008] = x[8,4096] (f32) · W[11008,4096]^T (f16) + bias (f32)
// Memory-bound (W = 90 MB, fits in 192 MB L2 on replay). Stream W with
// global_load_b128 into v_wmma_f32_16x16x32_f16; x staged in LDS as f16,
// zero-padded M 8->16, so the K-loop is branch- and convert-free.

typedef __attribute__((ext_vector_type(16))) _Float16 v16h;
typedef __attribute__((ext_vector_type(8)))  _Float16 v8h;
typedef __attribute__((ext_vector_type(4)))  _Float16 v4h;
typedef __attribute__((ext_vector_type(8)))  float    v8f;

#define IN_F   4096
#define OUT_F  11008
#define BATCH  8
#define KT     1024                 // K tile staged in LDS
#define LDS_STRIDE (KT + 8)         // +8 f16 pad: rows hit distinct bank groups
#define WAVES_PER_WG 4
#define BLOCK  (WAVES_PER_WG * 32)
#define NUM_WG (OUT_F / (16 * WAVES_PER_WG))   // 11008 / 64 = 172

__global__ __launch_bounds__(BLOCK)
void up_proj_wmma_f16(const float* __restrict__ x,
                      const _Float16* __restrict__ w,
                      const float* __restrict__ bias,
                      float* __restrict__ out)
{
    __shared__ _Float16 sx[16][LDS_STRIDE];   // 16 x 1032 f16 = 33 KB

    const int tid  = threadIdx.x;
    const int wave = tid >> 5;
    const int lane = tid & 31;
    const int n0   = (blockIdx.x * WAVES_PER_WG + wave) * 16;

    const int half = lane >> 4;     // K-half held by this lane
    const int m    = lane & 15;     // A row / D column index
    const int n    = n0 + m;        // output column held by this lane

    // B source: W[n][k .. k+15], 16 contiguous f16 = two global_load_b128
    const _Float16* __restrict__ wrow = w + (size_t)n * IN_F + half * 16;
    // A source (LDS): row m, chunks [kb+half*16, +8) and [kb+half*16+8, +8)
    const _Float16* arow = &sx[m][half * 16];

    // ---- one-time: zero entire LDS tile (covers pad rows 8..15 forever) ----
    {
        unsigned* z = (unsigned*)&sx[0][0];
        const int nwords = (16 * LDS_STRIDE) / 2;     // 8256
        for (int i = tid; i < nwords; i += BLOCK) z[i] = 0u;
    }

    v8f c0 = {};
    v8f c1 = {};

    for (int ko = 0; ko < IN_F; ko += KT) {
        __syncthreads();   // previous tile's reads done / zero-fill visible

        // ---- fill rows 0..7 of LDS with f16(x[ .. ][ko .. ko+KT)) ----
        #pragma unroll
        for (int it = 0; it < (BATCH * KT / 4) / BLOCK; ++it) {   // 16 iters
            const int wi = tid + it * BLOCK;
            const int mr = wi >> 8;           // 256 float4-chunks per row
            const int c4 = wi & 255;
            const float4 v = *(const float4*)(x + (size_t)mr * IN_F + ko + c4 * 4);
            v4h h;
            h[0] = (_Float16)v.x; h[1] = (_Float16)v.y;
            h[2] = (_Float16)v.z; h[3] = (_Float16)v.w;
            *(v4h*)&sx[mr][c4 * 4] = h;       // ds_store_b64
        }

        __syncthreads();

        // ---- K loop over this tile: branch-free, convert-free ----
        const _Float16* __restrict__ wt = wrow + ko;
        #pragma unroll 4
        for (int kb = 0; kb < KT; kb += 64) {
            // step 0: K = kb .. kb+31
            {
                v16h b = *(const v16h*)(wt + kb);                 // 2x global_load_b128
                v8h alo = *(const v8h*)(arow + kb);               // ds_load_b128
                v8h ahi = *(const v8h*)(arow + kb + 8);           // ds_load_b128
                v16h a = __builtin_shufflevector(alo, ahi,
                          0,1,2,3,4,5,6,7,8,9,10,11,12,13,14,15);
                c0 = __builtin_amdgcn_wmma_f32_16x16x32_f16(false, a, false, b,
                                                            (short)0, c0, false, false);
            }
            // step 1: K = kb+32 .. kb+63
            {
                v16h b = *(const v16h*)(wt + kb + 32);
                v8h alo = *(const v8h*)(arow + kb + 32);
                v8h ahi = *(const v8h*)(arow + kb + 40);
                v16h a = __builtin_shufflevector(alo, ahi,
                          0,1,2,3,4,5,6,7,8,9,10,11,12,13,14,15);
                c1 = __builtin_amdgcn_wmma_f32_16x16x32_f16(false, a, false, b,
                                                            (short)0, c1, false, false);
            }
        }
    }

    // D layout: VGPR r, lanes 0-15 -> (M=r, N=n0+lane). All 8 valid batch rows
    // live in lanes 0-15; lanes 16-31 hold padded rows (discarded).
    if (lane < 16) {
        const float bv = bias[n];
        #pragma unroll
        for (int r = 0; r < BATCH; ++r) {
            out[(size_t)r * OUT_F + n] = c0[r] + c1[r] + bv;
        }
    }
}

extern "C" void kernel_launch(void* const* d_in, const int* in_sizes, int n_in,
                              void* d_out, int out_size, void* d_ws, size_t ws_size,
                              hipStream_t stream) {
    (void)in_sizes; (void)n_in; (void)out_size; (void)d_ws; (void)ws_size;
    const float*    x    = (const float*)d_in[0];
    const _Float16* w    = (const _Float16*)d_in[1];
    const float*    bias = (const float*)d_in[2];
    float*          out  = (float*)d_out;

    dim3 grid(NUM_WG);     // 172 workgroups
    dim3 block(BLOCK);     // 128 threads = 4 wave32s (one per SIMD)
    hipLaunchKernelGGL(up_proj_wmma_f16, grid, block, 0, stream, x, w, bias, out);
}
